// NLLloss_batch_25220047962159
// MI455X (gfx1250) — compile-verified, matrix operand order checked by reading, not run on hardware
//
#include <hip/hip_runtime.h>
#include <hip/hip_bf16.h>

#define BB      16
#define KSLOT   7
#define CFEAT   256
#define KMK     10
#define HH      480
#define WW      480
#define HWN     (HH * WW)
#define ALPHA_W 0.5f
#define SIM_TH  0.75f
#define LN2F    0.69314718056f

typedef __attribute__((ext_vector_type(16))) _Float16 v16h;
typedef __attribute__((ext_vector_type(8)))  float    v8f;

// ---------------------------------------------------------------------------
// 0) zero workspace
// ---------------------------------------------------------------------------
__global__ void zero_kernel(float* ws, int n) {
    int i = blockIdx.x * blockDim.x + threadIdx.x;
    if (i < n) ws[i] = 0.0f;
}

// ---------------------------------------------------------------------------
// 1) row sums of mk (for the sums>0 filter) — float4 streaming
// ---------------------------------------------------------------------------
__global__ __launch_bounds__(256) void rowsum_kernel(const float* __restrict__ mk,
                                                     float* __restrict__ rowsum) {
    const int r = blockIdx.x;                 // b*KMK + row
    const float4* p = (const float4*)(mk + (size_t)r * HWN);
    float s = 0.0f;
    for (int i = threadIdx.x; i < HWN / 4; i += blockDim.x) {
        float4 v = p[i];
        s += v.x + v.y + v.z + v.w;
    }
    __shared__ float sm[256];
    sm[threadIdx.x] = s;
    __syncthreads();
    for (int off = 128; off > 0; off >>= 1) {
        if (threadIdx.x < off) sm[threadIdx.x] += sm[threadIdx.x + off];
        __syncthreads();
    }
    if (threadIdx.x == 0) rowsum[r] = sm[0];
}

// ---------------------------------------------------------------------------
// 2) select surviving mk rows (first min(count, KSLOT-1) rows with sum>0)
// ---------------------------------------------------------------------------
__global__ void select_kernel(const float* __restrict__ rowsum,
                              int* __restrict__ selidx, int* __restrict__ selcnt) {
    int b = threadIdx.x;
    if (b >= BB) return;
    int tmp[KMK];
    int cnt = 0;
    for (int i = 0; i < KMK; ++i)
        if (rowsum[b * KMK + i] > 0.0f) tmp[cnt++] = i;
    int C = (cnt >= KSLOT) ? (KSLOT - 1) : cnt;
    selcnt[b] = C;
    for (int c = 0; c < 8; ++c) selidx[b * 8 + c] = (c < C) ? tmp[c] : -1;
}

// ---------------------------------------------------------------------------
// 3) BCE cost matrix via WMMA.
//    Accumulates in LOG2 space: G2[c][k] = Sum_hw m*log2(a) + (1-m)*log2(1-a);
//    the uniform ln2 factor is applied once at the final atomicAdd, so the
//    inner loop logs are a single raw v_log_f32 (+ one 0/1-scale multiply).
//
//    A (16x32 f16) layout: lane L: M = L%16; element e holds
//        K = (e/8)*16 + (L/16)*8 + (e%8)
//    => per lane: two contiguous 8-float runs -> aligned b128 loads.
//    B (32x16 f16) mirror (N = L%16). C/D f32: lane L: N = L%16; VGPR v:
//    M = v + 8*(L/16). Padded lanes: safe pointers + 0/1 scale (no exec churn).
// ---------------------------------------------------------------------------
__global__ __launch_bounds__(32) void bce_cost_wmma_kernel(
    const float* __restrict__ alpha, const float* __restrict__ mk,
    const int* __restrict__ selidx, const int* __restrict__ selcnt,
    float* __restrict__ costws)
{
    const int b     = blockIdx.y;
    const int split = blockIdx.x;
    const int nsplt = gridDim.x;
    const int lane  = threadIdx.x;
    const int half  = lane >> 4;    // 0/1
    const int sub   = lane & 15;    // A: row index c ; B: col index k

    const int   C      = selcnt[b];
    const int   row    = (sub < C) ? selidx[b * 8 + sub] : 0;
    const float mscale = (sub < C) ? 1.0f : 0.0f;
    const int   k      = (sub < KSLOT) ? sub : 0;
    const float ascale = (sub < KSLOT) ? 1.0f : 0.0f;
    const float* __restrict__ mrow = mk    + ((size_t)(b * KMK)   + row) * HWN;
    const float* __restrict__ arow = alpha + ((size_t)(b * KSLOT) + k)   * HWN;

    v8f acc = {};
    const int nchunk = HWN / 64;  // 3600, exact
    for (int ch = split; ch < nchunk; ch += nsplt) {
        const int base = ch * 64;
#pragma unroll
        for (int s = 0; s < 2; ++s) {
            const int b0 = base + s * 32 + half * 8;   // 32B-aligned run start
            float4 m0 = *(const float4*)(mrow + b0);
            float4 m1 = *(const float4*)(mrow + b0 + 4);
            float4 m2 = *(const float4*)(mrow + b0 + 16);
            float4 m3 = *(const float4*)(mrow + b0 + 20);
            float4 a0 = *(const float4*)(arow + b0);
            float4 a1v = *(const float4*)(arow + b0 + 4);
            float4 a2v = *(const float4*)(arow + b0 + 16);
            float4 a3v = *(const float4*)(arow + b0 + 20);

            const float mv[16] = {m0.x, m0.y, m0.z, m0.w, m1.x, m1.y, m1.z, m1.w,
                                  m2.x, m2.y, m2.z, m2.w, m3.x, m3.y, m3.z, m3.w};
            const float av[16] = {a0.x, a0.y, a0.z, a0.w, a1v.x, a1v.y, a1v.z, a1v.w,
                                  a2v.x, a2v.y, a2v.z, a2v.w, a3v.x, a3v.y, a3v.z, a3v.w};

            v16h A1, A2, BLA, BL1A;
#pragma unroll
            for (int e = 0; e < 16; ++e) {
                float m = mscale * mv[e];
                A1[e] = (_Float16)m;
                A2[e] = (_Float16)(mscale - m);          // mscale*(1-mval)
                float a = fminf(fmaxf(av[e], 1e-6f), 1.0f - 1e-6f);
                BLA[e]  = (_Float16)(ascale * __log2f(a));        // log2 space
                BL1A[e] = (_Float16)(ascale * __log2f(1.0f - a)); // log2(1-a)
            }
            acc = __builtin_amdgcn_wmma_f32_16x16x32_f16(false, A1, false, BLA,
                                                         (short)0, acc, false, false);
            acc = __builtin_amdgcn_wmma_f32_16x16x32_f16(false, A2, false, BL1A,
                                                         (short)0, acc, false, false);
        }
    }
#pragma unroll
    for (int v = 0; v < 8; ++v) {
        const int M = v + (half << 3);
        atomicAdd(&costws[b * 256 + M * 16 + sub], LN2F * acc[v]);  // back to ln
    }
}

// ---------------------------------------------------------------------------
// 4) brute-force Hungarian (lexicographic permutations; strict < keeps the
//    lexicographically-first minimizer, matching itertools + "c < best")
// ---------------------------------------------------------------------------
__device__ __forceinline__ int perm_count(int n, int r) {
    int p = 1;
    for (int i = 0; i < r; ++i) p *= (n - i);
    return p;
}

__global__ __launch_bounds__(256) void hungarian_kernel(const float* __restrict__ costws,
                                                        const int* __restrict__ selcnt,
                                                        int* __restrict__ cols) {
    const int b = blockIdx.x;
    const int C = selcnt[b];
    __shared__ float scost[6 * KSLOT];
    if (threadIdx.x < 6 * KSLOT) {
        int c = threadIdx.x / KSLOT, k = threadIdx.x % KSLOT;
        scost[threadIdx.x] = -costws[b * 256 + c * 16 + k];  // cost ∝ -G'
    }
    __syncthreads();
    const int total = (C > 0) ? perm_count(KSLOT, C) : 0;
    float best = 3.4e38f;
    int bestIdx = -1;
    for (int idx = threadIdx.x; idx < total; idx += blockDim.x) {
        int rem = idx, used = 0;
        float s = 0.0f;
        for (int j = 0; j < C; ++j) {
            int f = perm_count(KSLOT - 1 - j, C - 1 - j);
            int d = rem / f; rem -= d * f;
            int col = 0;
            for (int q = 0; q < KSLOT; ++q) {
                if (!((used >> q) & 1)) {
                    if (d == 0) { col = q; break; }
                    --d;
                }
            }
            used |= 1 << col;
            s += scost[j * KSLOT + col];
        }
        if (s < best) { best = s; bestIdx = idx; }
    }
    __shared__ float rb[256];
    __shared__ int   ri[256];
    rb[threadIdx.x] = best; ri[threadIdx.x] = bestIdx;
    __syncthreads();
    for (int off = 128; off > 0; off >>= 1) {
        if (threadIdx.x < off) {
            float oc = rb[threadIdx.x + off];
            int   oi = ri[threadIdx.x + off];
            if (oi >= 0 && (ri[threadIdx.x] < 0 || oc < rb[threadIdx.x] ||
                            (oc == rb[threadIdx.x] && oi < ri[threadIdx.x]))) {
                rb[threadIdx.x] = oc; ri[threadIdx.x] = oi;
            }
        }
        __syncthreads();
    }
    if (threadIdx.x == 0) {
        for (int j = 0; j < 8; ++j) cols[b * 8 + j] = -1;
        int idx = (ri[0] < 0) ? 0 : ri[0];
        int rem = idx, used = 0;
        for (int j = 0; j < C; ++j) {
            int f = perm_count(KSLOT - 1 - j, C - 1 - j);
            int d = rem / f; rem -= d * f;
            int col = 0;
            for (int q = 0; q < KSLOT; ++q) {
                if (!((used >> q) & 1)) {
                    if (d == 0) { col = q; break; }
                    --d;
                }
            }
            used |= 1 << col;
            cols[b * 8 + j] = col;
        }
    }
}

// ---------------------------------------------------------------------------
// 5) cosine-sim selection: one thread per (b, slot); unmatched slots get
//    w_sel=1 if max cosine sim vs ALL matched features (all batches) < 0.75
// ---------------------------------------------------------------------------
__global__ __launch_bounds__(128) void wsel_kernel(const float* __restrict__ feat,
                                                   const int* __restrict__ cols,
                                                   const int* __restrict__ selcnt,
                                                   float* __restrict__ w_sel) {
    int t = blockIdx.x * blockDim.x + threadIdx.x;
    if (t >= BB * KSLOT) return;
    int b = t / KSLOT, slot = t % KSLOT;
    int C = selcnt[b];
    for (int c = 0; c < C; ++c)
        if (cols[b * 8 + c] == slot) return;  // matched slot -> w_sel stays 0
    const float* uf = feat + ((size_t)(b * KSLOT + slot)) * CFEAT;
    float un = 0.0f;
    for (int d = 0; d < CFEAT; ++d) un += uf[d] * uf[d];
    un = fmaxf(sqrtf(un), 1e-12f);
    float best = -3.4e38f;
    bool any = false;
    for (int b2 = 0; b2 < BB; ++b2) {
        int C2 = selcnt[b2];
        for (int c = 0; c < C2; ++c) {
            int ms = cols[b2 * 8 + c];
            const float* mf = feat + ((size_t)(b2 * KSLOT + ms)) * CFEAT;
            float dot = 0.0f, mn = 0.0f;
            for (int d = 0; d < CFEAT; ++d) { dot += mf[d] * uf[d]; mn += mf[d] * mf[d]; }
            mn = fmaxf(sqrtf(mn), 1e-12f);
            best = fmaxf(best, dot / (un * mn));
            any = true;
        }
    }
    if (!any || best < SIM_TH) w_sel[b * 8 + slot] = 1.0f;
}

// ---------------------------------------------------------------------------
// 6) final masked loss reduction (bandwidth-bound streaming pass, float4).
//    term1 accumulated in log2 space; -ln2 applied once per block.
// ---------------------------------------------------------------------------
__global__ __launch_bounds__(256) void loss_accum_kernel(
    const float* __restrict__ mfg, const float* __restrict__ Wfg,
    const float* __restrict__ inst, const float* __restrict__ w_sel,
    float* __restrict__ acc)
{
    const int b = blockIdx.y;
    float wsv[KSLOT];
#pragma unroll
    for (int k = 0; k < KSLOT; ++k) wsv[k] = w_sel[b * 8 + k];
    const float4* m4 = (const float4*)(mfg + (size_t)b * HWN);
    const float4* p4 = (const float4*)(Wfg + (size_t)b * HWN);
    const float*  ib = inst + (size_t)b * KSLOT * HWN;
    float t1l2 = 0.0f, nz = 0.0f, t2 = 0.0f;
    const int n4 = HWN / 4;
    for (int i = blockIdx.x * blockDim.x + threadIdx.x; i < n4;
         i += gridDim.x * blockDim.x) {
        float4 mv = m4[i];
        float4 pv = p4[i];
        float wf[4] = {0.0f, 0.0f, 0.0f, 0.0f};
#pragma unroll
        for (int k = 0; k < KSLOT; ++k) {
            float4 iv = *(const float4*)(ib + (size_t)k * HWN + i * 4);
            wf[0] += wsv[k] * iv.x; wf[1] += wsv[k] * iv.y;
            wf[2] += wsv[k] * iv.z; wf[3] += wsv[k] * iv.w;
        }
        const float mvv[4] = {mv.x, mv.y, mv.z, mv.w};
        const float pvv[4] = {pv.x, pv.y, pv.z, pv.w};
#pragma unroll
        for (int j = 0; j < 4; ++j) {
            t1l2 += mvv[j] * __log2f(pvv[j] + 1e-6f);   // log2; scaled later
            float z = (mvv[j] == 0.0f) ? 1.0f : 0.0f;
            nz += z;
            t2 += z * wf[j];
        }
    }
    __shared__ float s1[256], s2[256], s3[256];
    s1[threadIdx.x] = t1l2; s2[threadIdx.x] = nz; s3[threadIdx.x] = t2;
    __syncthreads();
    for (int off = 128; off > 0; off >>= 1) {
        if (threadIdx.x < off) {
            s1[threadIdx.x] += s1[threadIdx.x + off];
            s2[threadIdx.x] += s2[threadIdx.x + off];
            s3[threadIdx.x] += s3[threadIdx.x + off];
        }
        __syncthreads();
    }
    if (threadIdx.x == 0) {
        atomicAdd(&acc[b * 4 + 0], -LN2F * s1[0]);  // term1 numerator (ln space)
        atomicAdd(&acc[b * 4 + 1], s2[0]);
        atomicAdd(&acc[b * 4 + 2], s3[0]);
    }
}

__global__ void finalize_kernel(const float* __restrict__ acc, float* __restrict__ out) {
    if (threadIdx.x != 0 || blockIdx.x != 0) return;
    float s = 0.0f;
    for (int b = 0; b < BB; ++b) {
        float t1 = acc[b * 4 + 0] / (float)HWN;
        float ns = acc[b * 4 + 1];
        float t2 = (ns > 0.0f) ? (ALPHA_W * acc[b * 4 + 2] / ns) : 0.0f;
        s += t1 + t2;
    }
    out[0] = s / (float)BB;
}

// ---------------------------------------------------------------------------
// launch
// ---------------------------------------------------------------------------
extern "C" void kernel_launch(void* const* d_in, const int* in_sizes, int n_in,
                              void* d_out, int out_size, void* d_ws, size_t ws_size,
                              hipStream_t stream) {
    (void)in_sizes; (void)n_in; (void)out_size; (void)ws_size;
    const float* mfg   = (const float*)d_in[0];
    const float* Wfg   = (const float*)d_in[1];
    const float* feat  = (const float*)d_in[2];
    const float* alpha = (const float*)d_in[3];
    const float* mk    = (const float*)d_in[4];
    const float* inst  = (const float*)d_in[5];
    float* out = (float*)d_out;

    float* ws     = (float*)d_ws;
    float* rowsum = ws;                   // 160 floats
    int*   selcnt = (int*)(ws + 160);     // 16
    int*   selidx = (int*)(ws + 176);     // 128
    float* costws = ws + 304;             // 4096 (B x 16 x 16)
    int*   cols   = (int*)(ws + 4400);    // 128
    float* w_sel  = ws + 4528;            // 128 (B x 8)
    float* acc    = ws + 4656;            // 64  (B x 4)
    const int WS_FLOATS = 4720;

    zero_kernel<<<(WS_FLOATS + 255) / 256, 256, 0, stream>>>(ws, WS_FLOATS);
    rowsum_kernel<<<BB * KMK, 256, 0, stream>>>(mk, rowsum);
    select_kernel<<<1, 32, 0, stream>>>(rowsum, selidx, selcnt);
    dim3 gcost(32, BB);  // 512 waves streaming HW
    bce_cost_wmma_kernel<<<gcost, 32, 0, stream>>>(alpha, mk, selidx, selcnt, costws);
    hungarian_kernel<<<BB, 256, 0, stream>>>(costws, selcnt, cols);
    wsel_kernel<<<1, 128, 0, stream>>>(feat, cols, selcnt, w_sel);
    dim3 gloss(128, BB);
    loss_accum_kernel<<<gloss, 256, 0, stream>>>(mfg, Wfg, inst, w_sel, acc);
    finalize_kernel<<<1, 1, 0, stream>>>(acc, out);
}
